// SequenceModel_4148938408922
// MI455X (gfx1250) — matrix operand order repaired and checked by reading, not compiled, not run on hardware
//
#include <hip/hip_runtime.h>
#include <hip/hip_bf16.h>

typedef __attribute__((ext_vector_type(16))) __bf16 v16bf;
typedef __attribute__((ext_vector_type(8)))  float  v8f;

#define G_  8
#define H_  512
#define I_  1028
#define KP  1056          // I padded to multiple of 32
#define B_  128
#define T_  64
#define GH  (G_*H_)       // 4096
#define TH3 1536          // 3*H

static __device__ __forceinline__ unsigned short f2bf(float f) {
    unsigned u = __float_as_uint(f);
    u += 0x7FFFu + ((u >> 16) & 1u);      // round-to-nearest-even
    return (unsigned short)(u >> 16);
}

union FragBF { v16bf v; uint4 q[2]; };

static __device__ __forceinline__ v8f wmma_bf16(v16bf a, v16bf b, v8f c) {
    return __builtin_amdgcn_wmma_f32_16x16x32_bf16(
        /*neg_a=*/false, a, /*neg_b=*/false, b,
        /*c_mod=*/(short)0, c, /*reuse_a=*/false, /*reuse_b=*/false);
}

// ---- prep: x_bf[b*T+t][k] = concat(a,z) padded to KP, bf16 ----
__global__ void prep_x(const float* __restrict__ z, const float* __restrict__ a,
                       unsigned short* __restrict__ xbf) {
    int idx = blockIdx.x * 256 + threadIdx.x;
    const int N = B_ * T_ * KP;
    if (idx >= N) return;
    int row = idx / KP, k = idx % KP;     // row = b*T + t
    float v = 0.f;
    if (k < 4)            v = a[row * 4 + k];
    else if (k < I_)      v = z[row * 1024 + (k - 4)];
    xbf[idx] = f2bf(v);
}

// ---- prep: bf16 weights (W_ih padded to KP cols, W_hh as-is) ----
__global__ void prep_w(const float* __restrict__ Wih, const float* __restrict__ Whh,
                       unsigned short* __restrict__ wihb, unsigned short* __restrict__ whhb) {
    int idx = blockIdx.x * 256 + threadIdx.x;
    const int Nih = G_ * TH3 * KP;        // 12,976,128
    const int Nhh = G_ * TH3 * H_;        //  6,291,456
    if (idx < Nih) {
        int row = idx / KP, k = idx % KP;
        wihb[idx] = f2bf(k < I_ ? Wih[(size_t)row * I_ + k] : 0.f);
    } else if (idx < Nih + Nhh) {
        int j = idx - Nih;
        whhb[j] = f2bf(Whh[j]);
    }
}

// ---- prep: h0 (B, G*H) f32 -> hbf[g][b][u] bf16 ----
__global__ void init_h(const float* __restrict__ h0, unsigned short* __restrict__ hbf) {
    int idx = blockIdx.x * 256 + threadIdx.x;
    if (idx >= B_ * GH) return;
    int b = idx / GH, r = idx % GH, g = r / H_, u = r % H_;
    hbf[((size_t)g * B_ + b) * H_ + u] = f2bf(h0[idx]);
}

// ---- one GRU timestep, fused x-projection + recurrence, WMMA bf16 ----
__global__ __launch_bounds__(128)
void gru_step(const unsigned short* __restrict__ xbf,
              const unsigned short* __restrict__ wih,
              const unsigned short* __restrict__ whh,
              unsigned short* __restrict__ hbf,
              const float* __restrict__ b_ih, const float* __restrict__ b_hh,
              const float* __restrict__ hprev_base, int hprev_bstride,
              float* __restrict__ out, int t) {
    const int lane  = threadIdx.x & 31;
    const int wv    = threadIdx.x >> 5;
    const int g     = blockIdx.z;
    const int mtile = blockIdx.y;          // 4 tiles of 32 batch rows
    const int ntile = blockIdx.x;          // 8 tiles of 64 units
    const int col   = lane & 15;
    const int half  = lane >> 4;
    const int n0    = ntile * 64 + wv * 16;
    const int unit  = n0 + col;            // output unit in [0,512)
    const int b0    = mtile * 32;
    const int am    = lane & 15;           // A-fragment row (batch within tile)

    v8f zeroacc = {};
    v8f acc_r[2]  = {zeroacc, zeroacc};
    v8f acc_z[2]  = {zeroacc, zeroacc};
    v8f acc_xn[2] = {zeroacc, zeroacc};
    v8f acc_hn[2] = {zeroacc, zeroacc};

    // ---- phase 1: x(t) . W_ih  (K = 1056) ----
    const unsigned short* aptr[2];
    for (int f = 0; f < 2; ++f) {
        int b = b0 + f * 16 + am;
        aptr[f] = xbf + ((size_t)b * T_ + t) * KP + half * 8;
    }
    const unsigned short* bptr[3];
    for (int gt = 0; gt < 3; ++gt)
        bptr[gt] = wih + (size_t)(g * TH3 + gt * H_ + unit) * KP + half * 8;

    for (int kb = 0; kb < KP; kb += 32) {
        v16bf bfr[3];
        for (int gt = 0; gt < 3; ++gt) {
            FragBF fb;
            fb.q[0] = *(const uint4*)(bptr[gt] + kb);
            fb.q[1] = *(const uint4*)(bptr[gt] + kb + 16);
            bfr[gt] = fb.v;
        }
        for (int f = 0; f < 2; ++f) {
            FragBF fa;
            fa.q[0] = *(const uint4*)(aptr[f] + kb);
            fa.q[1] = *(const uint4*)(aptr[f] + kb + 16);
            v16bf av = fa.v;
            acc_r[f]  = wmma_bf16(av, bfr[0], acc_r[f]);
            acc_z[f]  = wmma_bf16(av, bfr[1], acc_z[f]);
            acc_xn[f] = wmma_bf16(av, bfr[2], acc_xn[f]);
        }
    }

    // ---- phase 2: h(t-1) . W_hh  (K = 512) ----
    const unsigned short* haptr[2];
    for (int f = 0; f < 2; ++f) {
        int b = b0 + f * 16 + am;
        haptr[f] = hbf + ((size_t)g * B_ + b) * H_ + half * 8;
    }
    const unsigned short* hbptr[3];
    for (int gt = 0; gt < 3; ++gt)
        hbptr[gt] = whh + (size_t)(g * TH3 + gt * H_ + unit) * H_ + half * 8;

    for (int kb = 0; kb < H_; kb += 32) {
        v16bf bfr[3];
        for (int gt = 0; gt < 3; ++gt) {
            FragBF fb;
            fb.q[0] = *(const uint4*)(hbptr[gt] + kb);
            fb.q[1] = *(const uint4*)(hbptr[gt] + kb + 16);
            bfr[gt] = fb.v;
        }
        for (int f = 0; f < 2; ++f) {
            FragBF fa;
            fa.q[0] = *(const uint4*)(haptr[f] + kb);
            fa.q[1] = *(const uint4*)(haptr[f] + kb + 16);
            v16bf av = fa.v;
            acc_r[f]  = wmma_bf16(av, bfr[0], acc_r[f]);
            acc_z[f]  = wmma_bf16(av, bfr[1], acc_z[f]);
            acc_hn[f] = wmma_bf16(av, bfr[2], acc_hn[f]);
        }
    }

    // ---- gates (f32) ----
    const float bir = b_ih[g * TH3 + unit],          bhr = b_hh[g * TH3 + unit];
    const float biz = b_ih[g * TH3 + H_ + unit],     bhz = b_hh[g * TH3 + H_ + unit];
    const float bin = b_ih[g * TH3 + 2 * H_ + unit], bhn = b_hh[g * TH3 + 2 * H_ + unit];

    for (int f = 0; f < 2; ++f) {
        for (int i = 0; i < 8; ++i) {
            int b = b0 + f * 16 + i + 8 * half;   // C row = vgpr + 8*half
            float r = 1.f / (1.f + __expf(-(acc_r[f][i] + bir + bhr)));
            float u = 1.f / (1.f + __expf(-(acc_z[f][i] + biz + bhz)));
            float n = tanhf(acc_xn[f][i] + bin + r * (acc_hn[f][i] + bhn));
            float hp = hprev_base[(size_t)b * hprev_bstride + g * H_ + unit];
            float hnew = (1.f - u) * n + u * hp;
            out[((size_t)b * T_ + t) * GH + g * H_ + unit] = hnew;
            hbf[((size_t)g * B_ + b) * H_ + unit] = f2bf(hnew);
        }
    }
}

extern "C" void kernel_launch(void* const* d_in, const int* in_sizes, int n_in,
                              void* d_out, int out_size, void* d_ws, size_t ws_size,
                              hipStream_t stream) {
    (void)in_sizes; (void)n_in; (void)out_size; (void)ws_size;
    const float* z   = (const float*)d_in[0];
    const float* a   = (const float*)d_in[1];
    const float* h0  = (const float*)d_in[2];
    const float* Wih = (const float*)d_in[3];
    const float* Whh = (const float*)d_in[4];
    const float* bih = (const float*)d_in[5];
    const float* bhh = (const float*)d_in[6];
    float* out = (float*)d_out;

    char* ws = (char*)d_ws;
    size_t off = 0;
    auto alloc = [&](size_t bytes) {
        void* p = ws + off;
        off = (off + bytes + 255) & ~(size_t)255;
        return p;
    };
    unsigned short* xbf  = (unsigned short*)alloc((size_t)B_ * T_ * KP * 2);   // 17.3 MB
    unsigned short* wihb = (unsigned short*)alloc((size_t)G_ * TH3 * KP * 2);  // 26.0 MB
    unsigned short* whhb = (unsigned short*)alloc((size_t)G_ * TH3 * H_ * 2);  // 12.6 MB
    unsigned short* hbf  = (unsigned short*)alloc((size_t)G_ * B_ * H_ * 2);   //  1.0 MB

    {
        int N = B_ * T_ * KP;
        prep_x<<<(N + 255) / 256, 256, 0, stream>>>(z, a, xbf);
    }
    {
        int N = G_ * TH3 * KP + G_ * TH3 * H_;
        prep_w<<<(N + 255) / 256, 256, 0, stream>>>(Wih, Whh, wihb, whhb);
    }
    {
        int N = B_ * GH;
        init_h<<<(N + 255) / 256, 256, 0, stream>>>(h0, hbf);
    }

    dim3 grid(8 /*N tiles*/, 4 /*M tiles*/, 8 /*blocks g*/);
    for (int t = 0; t < T_; ++t) {
        const float* hpb = (t == 0) ? h0 : (out + (size_t)(t - 1) * GH);
        int hstride = (t == 0) ? GH : T_ * GH;
        gru_step<<<grid, 128, 0, stream>>>(xbf, wihb, whhb, hbf, bih, bhh,
                                           hpb, hstride, out, t);
    }
}
